// GatedAttentionUnit_28054726377530
// MI455X (gfx1250) — compile-verified, matrix operand order checked by reading, not run on hardware
//
#include <hip/hip_runtime.h>
#include <hip/hip_bf16.h>

typedef __attribute__((ext_vector_type(16))) __bf16 v16bf;
typedef __attribute__((ext_vector_type(8)))  float  v8f;
typedef __attribute__((ext_vector_type(4)))  int    v4i;

typedef __attribute__((address_space(3))) __bf16   ls_bf16;   // explicit LDS pointee
typedef __attribute__((address_space(3))) unsigned ls_u32;
typedef __attribute__((address_space(3))) v4i      ls_v4i;
typedef __attribute__((address_space(1))) v4i      gl_v4i;

#define LSTR 40                 // LDS row stride in bf16 (80B: 16B-aligned rows, conflict-free frag loads)
#define TILE_ELEMS (128 * LSTR) // one 128x32 tile (padded) = 10KB

#if __has_builtin(__builtin_amdgcn_global_load_async_to_lds_b128)
#define HAS_ASYNC 1
#else
#define HAS_ASYNC 0
#endif

template <int N> __device__ __forceinline__ void wait_async() {
#if HAS_ASYNC
#if __has_builtin(__builtin_amdgcn_s_wait_asynccnt)
  __builtin_amdgcn_s_wait_asynccnt(N);
#else
  asm volatile("s_wait_asynccnt %0" ::"n"(N));
#endif
#endif
}

union FragU { v16bf v; unsigned u[8]; };

// 16x32 bf16 fragment from padded LDS tile. Lane 0-15 -> row r, K 0..7 and 16..23;
// lane 16-31 -> row r, K offset +8 (per CDNA5 ISA 16-bit A layout; B mirrors over columns).
// Each half is a contiguous 16B chunk -> compiler emits 2x ds_load_b128.
__device__ __forceinline__ v16bf ld_frag(const ls_bf16* lds, int rowBase, int lane) {
  int r = lane & 15, h = lane >> 4;
  const ls_bf16* p = lds + (rowBase + r) * LSTR + 8 * h;
  FragU f;
#pragma unroll
  for (int i = 0; i < 4; ++i) f.u[i]     = *(const ls_u32*)(p + 2 * i);
#pragma unroll
  for (int i = 0; i < 4; ++i) f.u[4 + i] = *(const ls_u32*)(p + 16 + 2 * i);
  return f.v;
}

// One K=32 step: wave (wm,wn) updates its 32x64 sub-tile (2x4 wmma tiles).
__device__ __forceinline__ void mma_step(const ls_bf16* lA, const ls_bf16* lB,
                                         int wm, int wn, int lane, v8f acc[2][4]) {
  v16bf af[2], bfv[4];
#pragma unroll
  for (int mi = 0; mi < 2; ++mi) af[mi] = ld_frag(lA, wm * 32 + mi * 16, lane);
#pragma unroll
  for (int ni = 0; ni < 4; ++ni) bfv[ni] = ld_frag(lB, wn * 64 + ni * 16, lane);
#pragma unroll
  for (int mi = 0; mi < 2; ++mi)
#pragma unroll
    for (int ni = 0; ni < 4; ++ni)
      acc[mi][ni] = __builtin_amdgcn_wmma_f32_16x16x32_bf16(
          false, af[mi], false, bfv[ni], (short)0, acc[mi][ni], false, false);
}

// Issue one 128x32 bf16 tile (row-major global, ld elems) into LDS.
// Async path: 2 x GLOBAL_LOAD_ASYNC_TO_LDS_B128 per wave (16B per lane), ASYNCcnt-tracked.
__device__ __forceinline__ void issue_tile(const __bf16* g, long row0, int col0, int ld,
                                           ls_bf16* lds, int tid) {
#pragma unroll
  for (int i = 0; i < 2; ++i) {
    int idx = tid + i * 256;
    int r = idx >> 2, c8 = (idx & 3) * 8;      // 4 x 16B chunks per 64B row
    const __bf16* gp = g + (row0 + r) * (long)ld + col0 + c8;
    ls_bf16* lp = lds + r * LSTR + c8;
#if HAS_ASYNC
    __builtin_amdgcn_global_load_async_to_lds_b128((gl_v4i*)gp, (ls_v4i*)lp, 0, 0);
#else
    *(ls_v4i*)lp = *(const v4i*)gp;
#endif
  }
}

// Double-buffered bf16 GEMM inner loop over K (multiple of 32).
// lds must hold 4 tiles: A0 A1 B0 B1.
__device__ __forceinline__ void gemm_loop(const __bf16* A, long am0, int lda,
                                          const __bf16* B, long bn0, int ldb, int K,
                                          ls_bf16* lds, int tid, int lane, int wm, int wn,
                                          v8f acc[2][4]) {
  ls_bf16* lA[2] = { lds, lds + TILE_ELEMS };
  ls_bf16* lB[2] = { lds + 2 * TILE_ELEMS, lds + 3 * TILE_ELEMS };
  issue_tile(A, am0, 0, lda, lA[0], tid);
  issue_tile(B, bn0, 0, ldb, lB[0], tid);
  int nsteps = K >> 5;
  for (int s = 0; s < nsteps; ++s) {
    int cur = s & 1;
    if (s + 1 < nsteps) {
      issue_tile(A, am0, (s + 1) * 32, lda, lA[1 - cur], tid);
      issue_tile(B, bn0, (s + 1) * 32, ldb, lB[1 - cur], tid);
      wait_async<4>();                 // 4 newest (next tile) may remain in flight
    } else {
      wait_async<0>();
    }
    __syncthreads();                   // all waves' current-tile chunks landed
    mma_step(lA[cur], lB[cur], wm, wn, lane, acc);
    __syncthreads();                   // safe to overwrite this buffer next round
  }
}

// ---------------- tiled transpose + f32->bf16 convert: dst[c][r] = src[r][c] ----------
__global__ __launch_bounds__(256) void k_tcvt(const float* src, __bf16* dst,
                                              int rows, int cols) {
  __shared__ float t[32][33];
  int c0 = blockIdx.x * 32, r0 = blockIdx.y * 32;
  int tx = threadIdx.x, ty = threadIdx.y;   // 32 x 8
#pragma unroll
  for (int i = 0; i < 4; ++i)
    t[ty + i * 8][tx] = src[(long)(r0 + ty + i * 8) * cols + c0 + tx];
  __syncthreads();
#pragma unroll
  for (int i = 0; i < 4; ++i)
    dst[(long)(c0 + ty + i * 8) * rows + r0 + tx] = (__bf16)t[tx][ty + i * 8];
}

// ---------------- hs f32 -> bf16 (vectorized) ----------------
__global__ __launch_bounds__(256) void k_cvt_hs(const float* hs, __bf16* hsb) {
  long i4 = (long)blockIdx.x * 256 + threadIdx.x;  // 2M float4s
  float4 v = ((const float4*)hs)[i4];
  union { __bf16 h[4]; uint2 u; } p;
  p.h[0] = (__bf16)v.x; p.h[1] = (__bf16)v.y; p.h[2] = (__bf16)v.z; p.h[3] = (__bf16)v.w;
  ((uint2*)hsb)[i4] = p.u;
}

// ---------------- gate: G = bf16(u * O) ----------------
__global__ __launch_bounds__(256) void k_gate(const float* u, const float* O, __bf16* G) {
  long i4 = (long)blockIdx.x * 256 + threadIdx.x;
  float4 a = ((const float4*)u)[i4];
  float4 b = ((const float4*)O)[i4];
  union { __bf16 h[4]; uint2 u2; } p;
  p.h[0] = (__bf16)(a.x * b.x); p.h[1] = (__bf16)(a.y * b.y);
  p.h[2] = (__bf16)(a.z * b.z); p.h[3] = (__bf16)(a.w * b.w);
  ((uint2*)G)[i4] = p.u2;
}

// ---------------- mask row sums (ll) ----------------
__global__ __launch_bounds__(256) void k_rowsum(const int* mask, float* ll) {
  __shared__ int red[256];
  long row = blockIdx.x;
  int t = threadIdx.x, s = 0;
  const int* p = mask + row * 2048;
  for (int i = t; i < 2048; i += 256) s += (p[i] != 0);
  red[t] = s; __syncthreads();
  for (int st = 128; st > 0; st >>= 1) { if (t < st) red[t] += red[t + st]; __syncthreads(); }
  if (t == 0) ll[row] = (float)red[0];
}

// ---------------- GEMM1: x = silu(hs @ Wi + bi); split u / vT / qk ----------------
__global__ __launch_bounds__(256) void k_gemm1(const __bf16* hsb, const __bf16* WiT,
                                               const float* bi, float* u, __bf16* vT,
                                               float* qk_raw) {
  __shared__ __align__(16) __bf16 smem[4 * TILE_ELEMS];
  ls_bf16* lds = (ls_bf16*)smem;
  int tid = threadIdx.x, lane = tid & 31, w = tid >> 5;
  int wm = w & 3, wn = w >> 2;
  long m0 = (long)blockIdx.x * 128; int n0 = blockIdx.y * 128;
  v8f acc[2][4] = {};
  gemm_loop(hsb, m0, 1024, WiT, n0, 1024, 1024, lds, tid, lane, wm, wn, acc);
  int nl = lane & 15, h = lane >> 4;
#pragma unroll
  for (int mi = 0; mi < 2; ++mi)
#pragma unroll
    for (int ni = 0; ni < 4; ++ni)
#pragma unroll
      for (int v = 0; v < 8; ++v) {
        long m = m0 + wm * 32 + mi * 16 + v + 8 * h;
        int  n = n0 + wn * 64 + ni * 16 + nl;
        float x = acc[mi][ni][v] + bi[n];
        float s = x * __builtin_amdgcn_rcpf(1.0f + __expf(-x));   // fast silu
        if (n < 1024) {
          u[m * 1024 + n] = s;
        } else if (n < 2048) {
          int b = (int)(m >> 11), ntok = (int)(m & 2047);
          vT[((long)b * 1024 + (n - 1024)) * 2048 + ntok] = (__bf16)s;
        } else {
          qk_raw[m * 128 + (n - 2048)] = s;
        }
      }
}

// ---------------- rotary + per-head affine; SCALE folded into q ----------------
__global__ __launch_bounds__(256) void k_rotary(const float* qk_raw, const float* sin_t,
                                                const float* cos_t, const float* qg,
                                                const float* qb, const float* kg,
                                                const float* kb, __bf16* q, __bf16* k) {
  int idx = blockIdx.x * 256 + threadIdx.x;       // 8192 * 64
  int tok = idx >> 6, c = idx & 63;
  int n = tok & 2047;
  float e = qk_raw[tok * 128 + 2 * c], o = qk_raw[tok * 128 + 2 * c + 1];
  float sn = sin_t[n * 64 + c], cs = cos_t[n * 64 + c];
  float qe = e * qg[2 * c] + qb[2 * c], qo = o * qg[2 * c + 1] + qb[2 * c + 1];
  float ke = e * kg[2 * c] + kb[2 * c], ko = o * kg[2 * c + 1] + kb[2 * c + 1];
  const float SC = 0.08838834764831845f;          // 1/sqrt(128)
  q[tok * 128 + c]      = (__bf16)((qe * cs - qo * sn) * SC);
  q[tok * 128 + c + 64] = (__bf16)((qe * sn + qo * cs) * SC);
  k[tok * 128 + c]      = (__bf16)(ke * cs - ko * sn);
  k[tok * 128 + c + 64] = (__bf16)(ke * sn + ko * cs);
}

// ---------------- scores: A = relu(mask ? q@kT : -inf)^2 / ll, bf16 ----------------
__global__ __launch_bounds__(256) void k_scores(const __bf16* q, const __bf16* k,
                                                const int* mask, const float* ll,
                                                __bf16* Amat) {
  __shared__ __align__(16) __bf16 smem[4 * TILE_ELEMS];
  ls_bf16* lds = (ls_bf16*)smem;
  int tid = threadIdx.x, lane = tid & 31, w = tid >> 5;
  int wm = w & 3, wn = w >> 2;
  int b = blockIdx.z;
  long m0 = (long)blockIdx.x * 128; int n0 = blockIdx.y * 128;
  v8f acc[2][4] = {};
  gemm_loop(q + (long)b * 2048 * 128, m0, 128,
            k + (long)b * 2048 * 128, n0, 128, 128, lds, tid, lane, wm, wn, acc);
  int nl = lane & 15, h = lane >> 4;
#pragma unroll
  for (int mi = 0; mi < 2; ++mi)
#pragma unroll
    for (int ni = 0; ni < 4; ++ni)
#pragma unroll
      for (int v = 0; v < 8; ++v) {
        long m = m0 + wm * 32 + mi * 16 + v + 8 * h;
        int  n = n0 + wn * 64 + ni * 16 + nl;
        float a = acc[mi][ni][v];
        int mk = mask[((long)b * 2048 + m) * 2048 + n];
        float r = (mk != 0 && a > 0.0f) ? a : 0.0f;
        Amat[((long)b * 2048 + m) * 2048 + n] = (__bf16)(r * r / ll[b * 2048 + m]);
      }
}

// ---------------- O = A @ v (per batch) ----------------
__global__ __launch_bounds__(256) void k_av(const __bf16* Amat, const __bf16* vT, float* O) {
  __shared__ __align__(16) __bf16 smem[4 * TILE_ELEMS];
  ls_bf16* lds = (ls_bf16*)smem;
  int tid = threadIdx.x, lane = tid & 31, w = tid >> 5;
  int wm = w & 3, wn = w >> 2;
  int b = blockIdx.z;
  long m0 = (long)blockIdx.x * 128; int n0 = blockIdx.y * 128;
  v8f acc[2][4] = {};
  gemm_loop(Amat + (long)b * 2048 * 2048, m0, 2048,
            vT + (long)b * 1024 * 2048, n0, 2048, 2048, lds, tid, lane, wm, wn, acc);
  int nl = lane & 15, h = lane >> 4;
#pragma unroll
  for (int mi = 0; mi < 2; ++mi)
#pragma unroll
    for (int ni = 0; ni < 4; ++ni)
#pragma unroll
      for (int v = 0; v < 8; ++v) {
        long m = m0 + wm * 32 + mi * 16 + v + 8 * h;
        int  n = n0 + wn * 64 + ni * 16 + nl;
        O[((long)b * 2048 + m) * 1024 + n] = acc[mi][ni][v];
      }
}

// ---------------- out = G @ Wo + bo  (G = bf16(u*O) precomputed) ----------------
__global__ __launch_bounds__(256) void k_out(const __bf16* G, const __bf16* WoT,
                                             const float* bo, float* out) {
  __shared__ __align__(16) __bf16 smem[4 * TILE_ELEMS];
  ls_bf16* lds = (ls_bf16*)smem;
  int tid = threadIdx.x, lane = tid & 31, w = tid >> 5;
  int wm = w & 3, wn = w >> 2;
  long m0 = (long)blockIdx.x * 128; int n0 = blockIdx.y * 128;
  v8f acc[2][4] = {};
  gemm_loop(G, m0, 1024, WoT, n0, 1024, 1024, lds, tid, lane, wm, wn, acc);
  int nl = lane & 15, h = lane >> 4;
#pragma unroll
  for (int mi = 0; mi < 2; ++mi)
#pragma unroll
    for (int ni = 0; ni < 4; ++ni)
#pragma unroll
      for (int v = 0; v < 8; ++v) {
        long m = m0 + wm * 32 + mi * 16 + v + 8 * h;
        int  n = n0 + wn * 64 + ni * 16 + nl;
        out[m * 1024 + n] = acc[mi][ni][v] + bo[n];
      }
}

extern "C" void kernel_launch(void* const* d_in, const int* in_sizes, int n_in,
                              void* d_out, int out_size, void* d_ws, size_t ws_size,
                              hipStream_t stream) {
  const float* hs    = (const float*)d_in[0];
  const int*   mask  = (const int*)  d_in[1];
  const float* sin_t = (const float*)d_in[2];
  const float* cos_t = (const float*)d_in[3];
  const float* Wi    = (const float*)d_in[4];
  const float* bi    = (const float*)d_in[5];
  const float* Wo    = (const float*)d_in[6];
  const float* bo    = (const float*)d_in[7];
  const float* qg    = (const float*)d_in[8];
  const float* qb    = (const float*)d_in[9];
  const float* kg    = (const float*)d_in[10];
  const float* kb    = (const float*)d_in[11];
  float* out = (float*)d_out;

  char* w = (char*)d_ws;
  size_t off = 0;
  auto alloc = [&](size_t bytes) {
    void* p = w + off;
    off += (bytes + 255) & ~(size_t)255;
    return p;
  };
  __bf16* hsb    = (__bf16*)alloc(8192ull * 1024 * 2);
  __bf16* WiT    = (__bf16*)alloc(2176ull * 1024 * 2);
  __bf16* WoT    = (__bf16*)alloc(1024ull * 1024 * 2);
  float*  u      = (float*) alloc(8192ull * 1024 * 4);
  __bf16* vT     = (__bf16*)alloc(4ull * 1024 * 2048 * 2);
  float*  qk_raw = (float*) alloc(8192ull * 128 * 4);
  __bf16* qbuf   = (__bf16*)alloc(8192ull * 128 * 2);
  __bf16* kbuf   = (__bf16*)alloc(8192ull * 128 * 2);
  float*  ll     = (float*) alloc(8192ull * 4);
  __bf16* Amat   = (__bf16*)alloc(4ull * 2048 * 2048 * 2);
  float*  O      = (float*) alloc(8192ull * 1024 * 4);
  __bf16* G      = (__bf16*)alloc(8192ull * 1024 * 2);

  k_tcvt   <<<dim3(68, 32), dim3(32, 8), 0, stream>>>(Wi, WiT, 1024, 2176);
  k_tcvt   <<<dim3(32, 32), dim3(32, 8), 0, stream>>>(Wo, WoT, 1024, 1024);
  k_cvt_hs <<<8192, 256, 0, stream>>>(hs, hsb);
  k_rowsum <<<8192, 256, 0, stream>>>(mask, ll);
  k_gemm1  <<<dim3(64, 17), 256, 0, stream>>>(hsb, WiT, bi, u, vT, qk_raw);
  k_rotary <<<2048, 256, 0, stream>>>(qk_raw, sin_t, cos_t, qg, qb, kg, kb, qbuf, kbuf);
  k_scores <<<dim3(16, 16, 4), 256, 0, stream>>>(qbuf, kbuf, mask, ll, Amat);
  k_av     <<<dim3(16, 8, 4), 256, 0, stream>>>(Amat, vT, O);
  k_gate   <<<8192, 256, 0, stream>>>(u, O, G);
  k_out    <<<dim3(64, 8), 256, 0, stream>>>(G, WoT, bo, out);
}
